// SelfAttention3_24970939859699
// MI455X (gfx1250) — matrix operand order repaired and checked by reading, not compiled, not run on hardware
//
#include <hip/hip_runtime.h>
#include <hip/hip_bf16.h>
#include <math.h>

// ---------------------------------------------------------------------------
// Problem constants (from reference)
// ---------------------------------------------------------------------------
#define BATCH 64
#define SEQ   512
#define HID   512
#define ATT   512
#define NCAT  4
#define VOC   1000
#define EDIM  128
#define NFEAT 6
#define NLAY  2
#define G3A   (3*ATT)          // 1536

typedef _Float16 v16h  __attribute__((ext_vector_type(16)));
typedef float    v8f   __attribute__((ext_vector_type(8)));
typedef float    f32x4 __attribute__((ext_vector_type(4)));
typedef _Float16 f16x4 __attribute__((ext_vector_type(4)));

// ---------------------------------------------------------------------------
// Positional encoding pe[s][d] (used once to build a lookup table)
// ---------------------------------------------------------------------------
__device__ __forceinline__ float pe_val(int s, int d) {
    float i   = (float)(d & ~1);
    float ang = (float)s / powf(10000.0f, i / (float)HID);
    return (d & 1) ? cosf(ang) : sinf(ang);
}

__global__ void build_pe(float* __restrict__ pe)
{
    int idx = blockIdx.x * blockDim.x + threadIdx.x;
    if (idx >= SEQ * HID) return;
    pe[idx] = pe_val(idx >> 9, idx & 511);
}

// ---------------------------------------------------------------------------
// Staging helpers for a 64x32 A tile and 32x128 B tile (256 threads).
// A: 2 float4/thread.  B: 4 float4/thread.  All global reads are b128.
// ---------------------------------------------------------------------------
template<int TRANSB>
__device__ __forceinline__ void stage_load(const float* __restrict__ A,
                                           const float* __restrict__ B,
                                           int K, int N, int mb, int nb, int kk,
                                           int t, f32x4 pa[2], f32x4 pb[4])
{
    #pragma unroll
    for (int e = 0; e < 2; ++e) {
        int i4 = t + e * 256;            // 0..511 float4s of the 64x32 A tile
        int r = i4 >> 3, c4 = i4 & 7;    // 8 float4 per row
        pa[e] = *(const f32x4*)(A + (size_t)(mb + r) * K + kk + c4 * 4);
    }
    if constexpr (TRANSB) {
        #pragma unroll
        for (int e = 0; e < 4; ++e) {
            int i4 = t + e * 256;        // 0..1023 float4s of 128x32 (B^T rows)
            int n = i4 >> 3, c4 = i4 & 7;            // along K, coalesced
            pb[e] = *(const f32x4*)(B + (size_t)(nb + n) * K + kk + c4 * 4);
        }
    } else {
        #pragma unroll
        for (int e = 0; e < 4; ++e) {
            int i4 = t + e * 256;        // 0..1023 float4s of 32x128
            int kr = i4 >> 5, n4 = i4 & 31;          // along N, coalesced
            pb[e] = *(const f32x4*)(B + (size_t)(kk + kr) * N + nb + n4 * 4);
        }
    }
}

template<int TRANSB>
__device__ __forceinline__ void stage_store(_Float16* __restrict__ As,
                                            _Float16* __restrict__ BsT,
                                            int t,
                                            const f32x4 pa[2], const f32x4 pb[4])
{
    #pragma unroll
    for (int e = 0; e < 2; ++e) {
        int i4 = t + e * 256;
        int r = i4 >> 3, c4 = i4 & 7;
        f16x4 h;
        #pragma unroll
        for (int j = 0; j < 4; ++j) h[j] = (_Float16)pa[e][j];
        *(f16x4*)(As + r * 32 + c4 * 4) = h;        // ds_store_b64
    }
    if constexpr (TRANSB) {
        #pragma unroll
        for (int e = 0; e < 4; ++e) {
            int i4 = t + e * 256;
            int n = i4 >> 3, c4 = i4 & 7;
            f16x4 h;
            #pragma unroll
            for (int j = 0; j < 4; ++j) h[j] = (_Float16)pb[e][j];
            *(f16x4*)(BsT + n * 32 + c4 * 4) = h;   // ds_store_b64
        }
    } else {
        #pragma unroll
        for (int e = 0; e < 4; ++e) {
            int i4 = t + e * 256;
            int kr = i4 >> 5, n4 = i4 & 31;
            #pragma unroll
            for (int j = 0; j < 4; ++j)             // transposed scatter
                BsT[(n4 * 4 + j) * 32 + kr] = (_Float16)pb[e][j];
        }
    }
}

// ---------------------------------------------------------------------------
// WMMA GEMM, templated on TRANSB:  C[M,N] = scale*(A x B[^T]) + bias + peTab
// Block tile 64x128: 8 wave32s; wave w owns row-stripe (w&3)*16 and four
// 16x16 col subtiles at (w>>2)*64 + {0,16,32,48}  ->  4 WMMAs / wave / K-step
// from 1 A fragment + 4 B fragments (all b128 LDS loads).
// Register-pipelined: tile k+1 global b128 loads issue before tile k's WMMAs.
// ---------------------------------------------------------------------------
template<int TRANSB>
__global__ __launch_bounds__(256)
void gemm_wmma_t(const float* __restrict__ A, const float* __restrict__ B,
                 float* __restrict__ C, int M, int N, int K,
                 long long sA, long long sB, long long sC,
                 const float* __restrict__ bias, float scale,
                 const float* __restrict__ peTab)
{
    A += (size_t)blockIdx.z * sA;
    B += (size_t)blockIdx.z * sB;
    C += (size_t)blockIdx.z * sC;

    const int nb = blockIdx.x * 128;
    const int mb = blockIdx.y * 64;

    __shared__ _Float16 As[64 * 32];      // [m][k]
    __shared__ _Float16 BsT[128 * 32];    // [n][k] (K-contiguous -> b128 frags)

    const int t    = threadIdx.x;
    const int w    = t >> 5;
    const int lane = t & 31;
    const int lo   = lane & 15;
    const int hi   = lane >> 4;

    const int mi = (w & 3) * 16;
    const int ng = (w >> 2) * 64;

    v8f c0 = {}, c1 = {}, c2 = {}, c3 = {};

    f32x4 pa[2], pb[4];
    stage_load<TRANSB>(A, B, K, N, mb, nb, 0, t, pa, pb);

    for (int kk = 0; kk < K; kk += 32) {
        stage_store<TRANSB>(As, BsT, t, pa, pb);
        __syncthreads();

        const bool more = (kk + 32) < K;
        if (more) {
            stage_load<TRANSB>(A, B, K, N, mb, nb, kk + 32, t, pa, pb);
            if (kk + 64 < K) {            // L2 prefetch two tiles ahead
                __builtin_prefetch(A + (size_t)(mb + (t >> 3)) * K + kk + 64
                                     + (t & 7) * 4, 0, 1);
                if constexpr (TRANSB)
                    __builtin_prefetch(B + (size_t)(nb + (t >> 3)) * K + kk + 64
                                         + (t & 7) * 4, 0, 1);
                else
                    __builtin_prefetch(B + (size_t)(kk + 64 + (t >> 5)) * N + nb
                                         + (t & 31) * 4, 0, 1);
            }
        }

        // ---- fragments: contiguous k = 16*hi + s in both A and B ----
        v16h a, b0, b1, b2, b3;
        #pragma unroll
        for (int v = 0; v < 8; ++v) {
            int ka = 2 * v + 16 * hi;                        // A 16x32 f16
            a[2 * v]     = As[(mi + lo) * 32 + ka];
            a[2 * v + 1] = As[(mi + lo) * 32 + ka + 1];
            int kb = 2 * v + 16 * hi;                        // B 32x16
            b0[2 * v]     = BsT[(ng +  0 + lo) * 32 + kb];
            b0[2 * v + 1] = BsT[(ng +  0 + lo) * 32 + kb + 1];
            b1[2 * v]     = BsT[(ng + 16 + lo) * 32 + kb];
            b1[2 * v + 1] = BsT[(ng + 16 + lo) * 32 + kb + 1];
            b2[2 * v]     = BsT[(ng + 32 + lo) * 32 + kb];
            b2[2 * v + 1] = BsT[(ng + 32 + lo) * 32 + kb + 1];
            b3[2 * v]     = BsT[(ng + 48 + lo) * 32 + kb];
            b3[2 * v + 1] = BsT[(ng + 48 + lo) * 32 + kb + 1];
        }

        c0 = __builtin_amdgcn_wmma_f32_16x16x32_f16(false, a, false, b0,
                                                    (short)0, c0, false, false);
        c1 = __builtin_amdgcn_wmma_f32_16x16x32_f16(false, a, false, b1,
                                                    (short)0, c1, false, false);
        c2 = __builtin_amdgcn_wmma_f32_16x16x32_f16(false, a, false, b2,
                                                    (short)0, c2, false, false);
        c3 = __builtin_amdgcn_wmma_f32_16x16x32_f16(false, a, false, b3,
                                                    (short)0, c3, false, false);
        __syncthreads();
    }

    // ---- epilogue: scale, bias, PE table, store f32 ----
    #pragma unroll
    for (int i = 0; i < 8; ++i) {
        int row = mb + mi + i + 8 * hi;
        const float* pr = peTab ? peTab + (size_t)(row & (SEQ - 1)) * HID : nullptr;
        float vq[4] = { c0[i], c1[i], c2[i], c3[i] };
        #pragma unroll
        for (int q = 0; q < 4; ++q) {
            int col = nb + ng + q * 16 + lo;
            float v = vq[q] * scale;
            if (bias) v += bias[col];
            if (pr)   v += pr[col];
            C[(size_t)row * N + col] = v;
        }
    }
}

// ---------------------------------------------------------------------------
// Build concatenated GEMM input xin[B*S, 1024] = [cat-embeddings | cont-proj]
// ---------------------------------------------------------------------------
__global__ void build_xin(const int* __restrict__ cate,
                          const float* __restrict__ cont,
                          const float* __restrict__ emb,
                          const float* __restrict__ Wc,
                          const float* __restrict__ bc,
                          float* __restrict__ xin)
{
    int idx = blockIdx.x * blockDim.x + threadIdx.x;
    if (idx >= BATCH * SEQ * 1024) return;
    int row = idx >> 10;
    int col = idx & 1023;
    float v;
    if (col < NCAT * EDIM) {
        int c = col >> 7;
        int e = col & 127;
        int id = cate[row * NCAT + c];
        v = emb[((size_t)c * VOC + id) * EDIM + e];
    } else {
        int j = col - NCAT * EDIM;
        v = bc[j];
        const float* cx = cont + (size_t)row * NFEAT;
        #pragma unroll
        for (int f = 0; f < NFEAT; ++f) v += cx[f] * Wc[f * HID + j];
    }
    xin[idx] = v;
}

// ---------------------------------------------------------------------------
// Online softmax over the QUERY axis (dim=1 of [B,S,S]) — column softmax.
// ---------------------------------------------------------------------------
__global__ void softmax_q(float* __restrict__ sc)
{
    int idx = blockIdx.x * blockDim.x + threadIdx.x;
    if (idx >= BATCH * SEQ) return;
    int b = idx >> 9, k = idx & 511;
    float* base = sc + (size_t)b * SEQ * SEQ + k;
    float m = -1e30f, s = 0.f;
    for (int q = 0; q < SEQ; ++q) {               // single online pass
        float v = base[(size_t)q * SEQ];
        if (v > m) { s = s * expf(m - v); m = v; }
        s += expf(v - m);
    }
    float inv = 1.f / s;
    for (int q = 0; q < SEQ; ++q)
        base[(size_t)q * SEQ] = expf(base[(size_t)q * SEQ] - m) * inv;
}

// ---------------------------------------------------------------------------
// GRU gate fusion for one timestep (PyTorch gate order r,z,n).
// ---------------------------------------------------------------------------
__global__ void gru_gate(const float* __restrict__ gi,
                         const float* __restrict__ gh,
                         float* __restrict__ h,
                         float* __restrict__ hs, int t)
{
    int idx = blockIdx.x * blockDim.x + threadIdx.x;
    if (idx >= BATCH * ATT) return;
    int b = idx >> 9, j = idx & 511;
    size_t gir = ((size_t)b * SEQ + t) * G3A;
    size_t ghr = (size_t)b * G3A;
    float ir  = gi[gir + j],            hr = gh[ghr + j];
    float iz  = gi[gir + ATT + j],      hz = gh[ghr + ATT + j];
    float inn = gi[gir + 2 * ATT + j],  hn = gh[ghr + 2 * ATT + j];
    float r  = 1.f / (1.f + expf(-(ir + hr)));
    float zg = 1.f / (1.f + expf(-(iz + hz)));
    float n  = tanhf(inn + r * hn);
    float hp = h[(size_t)b * ATT + j];
    float hv = (1.f - zg) * n + zg * hp;
    h[(size_t)b * ATT + j] = hv;
    hs[((size_t)b * SEQ + t) * ATT + j] = hv;
}

__global__ void zero_f32(float* __restrict__ p, int n)
{
    int i = blockIdx.x * blockDim.x + threadIdx.x;
    if (i < n) p[i] = 0.f;
}

// ---------------------------------------------------------------------------
// Final projection: out[row] = sigmoid(dot(hs[row,:], Wf) + bf). One wave/row.
// ---------------------------------------------------------------------------
__global__ void final_proj(const float* __restrict__ hs,
                           const float* __restrict__ Wf,
                           const float* __restrict__ bf,
                           float* __restrict__ out, int rows)
{
    int gid  = blockIdx.x * blockDim.x + threadIdx.x;
    int wid  = gid >> 5;
    int lane = gid & 31;
    if (wid >= rows) return;
    const float* r = hs + (size_t)wid * ATT;
    float acc = 0.f;
    for (int j = lane; j < ATT; j += 32) acc += r[j] * Wf[j];
    #pragma unroll
    for (int off = 16; off > 0; off >>= 1) acc += __shfl_down(acc, off, 32);
    if (lane == 0) out[wid] = 1.f / (1.f + expf(-(acc + bf[0])));
}

// ---------------------------------------------------------------------------
// Host-side pipeline
// ---------------------------------------------------------------------------
static inline void launch_gemm(const float* A, const float* B, float* C,
                               int M, int N, int K,
                               long long sA, long long sB, long long sC, int nbatch,
                               int transB, const float* bias, float scale,
                               const float* peTab, hipStream_t stream)
{
    dim3 grid(N / 128, M / 64, nbatch);
    if (transB)
        gemm_wmma_t<1><<<grid, 256, 0, stream>>>(A, B, C, M, N, K, sA, sB, sC,
                                                 bias, scale, peTab);
    else
        gemm_wmma_t<0><<<grid, 256, 0, stream>>>(A, B, C, M, N, K, sA, sB, sC,
                                                 bias, scale, peTab);
}

extern "C" void kernel_launch(void* const* d_in, const int* in_sizes, int n_in,
                              void* d_out, int out_size, void* d_ws, size_t ws_size,
                              hipStream_t stream)
{
    const int*   cate   = (const int*)  d_in[0];
    const float* cont   = (const float*)d_in[1];
    /* mask d_in[2], targets d_in[3] unused by the reference computation */
    const float* emb    = (const float*)d_in[4];
    const float* W_cont = (const float*)d_in[5];
    const float* b_cont = (const float*)d_in[6];
    const float* W_comb = (const float*)d_in[7];
    const float* b_comb = (const float*)d_in[8];
    const float* Wq     = (const float*)d_in[9];
    const float* Wk     = (const float*)d_in[10];
    const float* Wv     = (const float*)d_in[11];
    const float* Wi     = (const float*)d_in[12];   // [L,3A,A]
    const float* Wh     = (const float*)d_in[13];   // [L,3A,A]
    const float* bi     = (const float*)d_in[14];   // [L,3A]
    const float* bh     = (const float*)d_in[15];   // [L,3A]
    const float* W_fin  = (const float*)d_in[16];
    const float* b_fin  = (const float*)d_in[17];
    float* out = (float*)d_out;

    float* ws = (float*)d_ws;
    const size_t U = 16777216ULL;           // floats per 64MB region (B*S*512)
    const int ROWS = BATCH * SEQ;           // 32768

    // Region plan (phase-overlap safe):
    float* x      = ws + 0 * U;             // [ROWS,512]
    float* Q      = ws + 1 * U;
    float* Km     = ws + 2 * U;
    float* Vm     = ws + 3 * U;
    float* scores = ws + 4 * U;             // [B,512,512]
    float* z      = ws + 5 * U;             // [ROWS,512]
    float* xin    = ws + 4 * U;             // [ROWS,1024], dead before scores/z
    float* peTab  = ws + 1 * U;             // [512,512], dead before Q written
    // GRU phase (x/Q/K/V/scores dead):
    float* gi  = ws + 0 * U;                // [ROWS,1536]
    float* gh  = ws + 3 * U;                // [64,1536]
    float* h   = ws + 3 * U + 131072;       // [64,512]
    float* hs1 = ws + 4 * U;                // layer-1 hidden states
    float* hs2 = ws + 5 * U;                // layer-2 hidden states (z dead)

    const float inv_sqrt_a = 0.044194173824159216f;  // 1/sqrt(512)

    // 1) concat input + PE table
    build_xin<<<(ROWS * 1024 + 255) / 256, 256, 0, stream>>>(cate, cont, emb,
                                                             W_cont, b_cont, xin);
    build_pe<<<(SEQ * HID + 255) / 256, 256, 0, stream>>>(peTab);

    // 2) combine + bias + positional encoding
    launch_gemm(xin, W_comb, x, ROWS, HID, 1024, 0, 0, 0, 1,
                0, b_comb, 1.0f, peTab, stream);
    // 3) Q/K/V (Q overwrites the PE table region — PE no longer needed)
    launch_gemm(x, Wq, Q,  ROWS, ATT, HID, 0, 0, 0, 1, 0, nullptr, 1.0f, nullptr, stream);
    launch_gemm(x, Wk, Km, ROWS, ATT, HID, 0, 0, 0, 1, 0, nullptr, 1.0f, nullptr, stream);
    launch_gemm(x, Wv, Vm, ROWS, ATT, HID, 0, 0, 0, 1, 0, nullptr, 1.0f, nullptr, stream);
    // 4) scores = Q K^T / sqrt(A)  (batched over B, B-transposed layout)
    launch_gemm(Q, Km, scores, SEQ, SEQ, ATT,
                (long long)SEQ * ATT, (long long)SEQ * ATT, (long long)SEQ * SEQ,
                BATCH, 1, nullptr, inv_sqrt_a, nullptr, stream);
    // 5) softmax over query axis
    softmax_q<<<(BATCH * SEQ + 255) / 256, 256, 0, stream>>>(scores);
    // 6) z = scores @ V (batched)
    launch_gemm(scores, Vm, z, SEQ, ATT, SEQ,
                (long long)SEQ * SEQ, (long long)SEQ * ATT, (long long)SEQ * ATT,
                BATCH, 0, nullptr, 1.0f, nullptr, stream);

    // 7) two GRU layers (sequential recurrence; Wh stays resident in 192MB L2)
    const float* seq_in  = z;
    float*       seq_out = hs1;
    for (int l = 0; l < NLAY; ++l) {
        const float* Wil = Wi + (size_t)l * G3A * ATT;
        const float* Whl = Wh + (size_t)l * G3A * ATT;
        const float* bil = bi + (size_t)l * G3A;
        const float* bhl = bh + (size_t)l * G3A;
        // input gates for all timesteps at once
        launch_gemm(seq_in, Wil, gi, ROWS, G3A, ATT, 0, 0, 0, 1,
                    1, bil, 1.0f, nullptr, stream);
        zero_f32<<<(BATCH * ATT + 255) / 256, 256, 0, stream>>>(h, BATCH * ATT);
        for (int t = 0; t < SEQ; ++t) {
            launch_gemm(h, Whl, gh, BATCH, G3A, ATT, 0, 0, 0, 1,
                        1, bhl, 1.0f, nullptr, stream);
            gru_gate<<<(BATCH * ATT + 255) / 256, 256, 0, stream>>>(gi, gh, h,
                                                                    seq_out, t);
        }
        seq_in  = seq_out;                  // layer 2 consumes hs1
        seq_out = hs2;
    }

    // 8) final sigmoid projection (layer-2 output lives in hs2)
    final_proj<<<(ROWS * 32 + 255) / 256, 256, 0, stream>>>(hs2, W_fin, b_fin,
                                                            out, ROWS);
}
